// GraphUNetCoarsening_77386720739503
// MI455X (gfx1250) — compile-verified, matrix operand order checked by reading, not run on hardware
//
#include <hip/hip_runtime.h>
#include <hip/hip_bf16.h>

typedef __attribute__((ext_vector_type(16))) __bf16 v16bf;
typedef __attribute__((ext_vector_type(8)))  float  v8f;

// ---------------- problem constants (fixed by the reference) ----------------
constexpr int BATCH = 8;
constexpr int NNODE = 8192;
constexpr int SENT  = 64;
constexpr int CIN   = 256;
constexpr int HID   = 256;
constexpr int HID2  = 128;
constexpr int EPER  = 32768;
constexpr int NTOT  = BATCH * NNODE;      // 65536
constexpr int ETOT  = BATCH * EPER;       // 262144
constexpr int KSENT = 3;
constexpr int OTHERN = NNODE - SENT;      // 8128
constexpr int KOTHER = 4064;              // ceil(0.5 * 8128)
constexpr int PTOT  = BATCH * KSENT + BATCH * KOTHER;  // 32536
constexpr float SLOPE = 0.01f;

// ---------------- d_out layout (flat f32, tuple order) ----------------------
constexpr long O_X  = 0;                              // x_out        32536*256
constexpr long O_EI = (long)PTOT * CIN;               // edge_index   2*ETOT
constexpr long O_EA = O_EI + 2L * ETOT;               // edge_attr    ETOT
constexpr long O_EM = O_EA + ETOT;                    // emask        ETOT
constexpr long O_BO = O_EM + ETOT;                    // batch_out    PTOT
constexpr long O_PM = O_BO + PTOT;                    // perm         PTOT
constexpr long O_SP = O_PM + PTOT;                    // score[perm]  PTOT
constexpr long O_SS = O_SP + PTOT;                    // sentence_scores 512
constexpr long O_SB = O_SS + (long)BATCH * SENT;      // sentence_batch  512

// ---------------- helpers ---------------------------------------------------
// round-to-nearest-even bias for f32 -> bf16 truncation
static __device__ __forceinline__ unsigned int bfround(float f) {
    unsigned int u = __float_as_uint(f);
    return u + 0x7FFFu + ((u >> 16) & 1u);
}

// pack two f32 -> two bf16 in one dword: 2x(bfe+add3) + v_perm_b32
static __device__ __forceinline__ unsigned int pkbf(float lo, float hi) {
    return __builtin_amdgcn_perm(bfround(hi), bfround(lo), 0x07060302u);
}

union ABFrag { v16bf v; uint4 q[2]; };

// ---------------- bf16 WMMA GEMM:  Y[M,N] = X[M,K] @ W[K,N] -----------------
// block = 256 threads (8 wave32), block tile 128x64, wave tile 32x32 (2x2 WMMA)
// LDS double-buffered: global loads for slab k+1 overlap the WMMAs of slab k.
__global__ __launch_bounds__(256) void gemm_bf16_wmma(
    const float* __restrict__ X, const float* __restrict__ W,
    float* __restrict__ Y, int M, int N, int K)
{
    __shared__ __align__(16) unsigned short As[2][128][40];  // rows: 80B stride
    __shared__ __align__(16) unsigned short Bs[2][64][40];   // W tile, transposed

    const int tid   = threadIdx.x;
    const int lane  = tid & 31;
    const int wv    = tid >> 5;        // wave 0..7
    const int wm    = wv & 3;          // 4 waves along M (32 rows each)
    const int wn    = wv >> 2;         // 2 waves along N (32 cols each)
    const int lrow  = lane & 15;
    const int lhalf = lane >> 4;

    const int mBlk = blockIdx.x * 128;
    const int nBlk = blockIdx.y * 64;

    // fixed staging coordinates for this thread
    int arow[4], acol[4], bkk[2], bnn[2];
#pragma unroll
    for (int v = 0; v < 4; ++v) { int idx = (tid + v * 256) * 4; arow[v] = idx >> 5; acol[v] = idx & 31; }
#pragma unroll
    for (int v = 0; v < 2; ++v) { int idx = (tid + v * 256) * 4; bkk[v] = idx >> 6; bnn[v] = idx & 63; }

    v8f acc[2][2];
#pragma unroll
    for (int a = 0; a < 2; ++a)
#pragma unroll
        for (int b = 0; b < 2; ++b)
#pragma unroll
            for (int v = 0; v < 8; ++v) acc[a][b][v] = 0.0f;

    const int nK = K / 32;
    float4 fa[4], fb[2];

    // ---- prologue: stage slab 0 into LDS stage 0 ----
#pragma unroll
    for (int v = 0; v < 4; ++v)
        fa[v] = *(const float4*)(X + (long)(mBlk + arow[v]) * K + acol[v]);
#pragma unroll
    for (int v = 0; v < 2; ++v)
        fb[v] = *(const float4*)(W + (long)bkk[v] * N + nBlk + bnn[v]);
#pragma unroll
    for (int v = 0; v < 4; ++v) {
        uint2 p; p.x = pkbf(fa[v].x, fa[v].y); p.y = pkbf(fa[v].z, fa[v].w);
        *(uint2*)&As[0][arow[v]][acol[v]] = p;
    }
#pragma unroll
    for (int v = 0; v < 2; ++v) {
        unsigned int p01 = pkbf(fb[v].x, fb[v].y);
        unsigned int p23 = pkbf(fb[v].z, fb[v].w);
        Bs[0][bnn[v] + 0][bkk[v]] = (unsigned short)p01;
        Bs[0][bnn[v] + 1][bkk[v]] = (unsigned short)(p01 >> 16);
        Bs[0][bnn[v] + 2][bkk[v]] = (unsigned short)p23;
        Bs[0][bnn[v] + 3][bkk[v]] = (unsigned short)(p23 >> 16);
    }
    __syncthreads();

    for (int kt = 0; kt < nK; ++kt) {
        const int cur = kt & 1;
        const int nxt = cur ^ 1;

        // ---- issue global loads for slab kt+1 (overlap with WMMA below) ----
        if (kt + 1 < nK) {
            const int k0 = (kt + 1) * 32;
#pragma unroll
            for (int v = 0; v < 4; ++v)
                fa[v] = *(const float4*)(X + (long)(mBlk + arow[v]) * K + k0 + acol[v]);
#pragma unroll
            for (int v = 0; v < 2; ++v)
                fb[v] = *(const float4*)(W + (long)(k0 + bkk[v]) * N + nBlk + bnn[v]);
        }
        if (kt + 2 < nK)   // warm L2 for slab kt+2 (global_prefetch_b8)
            __builtin_prefetch(X + (long)(mBlk + (tid >> 1)) * K + (kt + 2) * 32, 0, 1);

        // ---- fragments from stage `cur` (two aligned b128 LDS loads each) ----
        ABFrag a[2], b[2];
#pragma unroll
        for (int t = 0; t < 2; ++t) {
            int row = wm * 32 + t * 16 + lrow;
            a[t].q[0] = *(const uint4*)&As[cur][row][lhalf * 8];
            a[t].q[1] = *(const uint4*)&As[cur][row][16 + lhalf * 8];
            int col = wn * 32 + t * 16 + lrow;
            b[t].q[0] = *(const uint4*)&Bs[cur][col][lhalf * 8];
            b[t].q[1] = *(const uint4*)&Bs[cur][col][16 + lhalf * 8];
        }
        // ---- 4 WMMAs per wave per K-slab ----
#pragma unroll
        for (int tm = 0; tm < 2; ++tm)
#pragma unroll
            for (int tn = 0; tn < 2; ++tn)
                acc[tm][tn] = __builtin_amdgcn_wmma_f32_16x16x32_bf16(
                    false, a[tm].v, false, b[tn].v,
                    (short)0, acc[tm][tn], false, false);

        __syncthreads();   // all waves done reading stage `cur` / prior `nxt`

        // ---- convert + store slab kt+1 into stage `nxt` ----
        if (kt + 1 < nK) {
#pragma unroll
            for (int v = 0; v < 4; ++v) {
                uint2 p; p.x = pkbf(fa[v].x, fa[v].y); p.y = pkbf(fa[v].z, fa[v].w);
                *(uint2*)&As[nxt][arow[v]][acol[v]] = p;
            }
#pragma unroll
            for (int v = 0; v < 2; ++v) {
                unsigned int p01 = pkbf(fb[v].x, fb[v].y);
                unsigned int p23 = pkbf(fb[v].z, fb[v].w);
                Bs[nxt][bnn[v] + 0][bkk[v]] = (unsigned short)p01;
                Bs[nxt][bnn[v] + 1][bkk[v]] = (unsigned short)(p01 >> 16);
                Bs[nxt][bnn[v] + 2][bkk[v]] = (unsigned short)p23;
                Bs[nxt][bnn[v] + 3][bkk[v]] = (unsigned short)(p23 >> 16);
            }
        }
        __syncthreads();
    }

    // ---- store C/D (16x16 f32 layout: lane = N, vgpr v -> M = v + 8*half) ----
#pragma unroll
    for (int tm = 0; tm < 2; ++tm)
#pragma unroll
        for (int tn = 0; tn < 2; ++tn) {
            int gc = nBlk + wn * 32 + tn * 16 + lrow;
#pragma unroll
            for (int v = 0; v < 8; ++v) {
                int gr = mBlk + wm * 32 + tm * 16 + v + 8 * lhalf;
                Y[(long)gr * N + gc] = acc[tm][tn][v];
            }
        }
}

// ---------------- normalization / scatter kernels ---------------------------
__global__ void zero_kernel(float* p, int n) {
    int i = blockIdx.x * 256 + threadIdx.x;
    if (i < n) p[i] = 0.0f;
}

__global__ void deg_scatter_kernel(const int* __restrict__ dst,
                                   const float* __restrict__ ew,
                                   float* __restrict__ deg) {
    int e = blockIdx.x * 256 + threadIdx.x;
    if (e < ETOT) atomicAdd(&deg[dst[e]], ew[e]);
}

__global__ void dis_kernel(float* d) {
    int i = blockIdx.x * 256 + threadIdx.x;
    if (i < NTOT) d[i] = rsqrtf(d[i] + 1.0f);
}

__global__ void coef_kernel(const int* __restrict__ src, const int* __restrict__ dst,
                            const float* __restrict__ ew, const float* __restrict__ dis,
                            float* __restrict__ coef) {
    int e = blockIdx.x * 256 + threadIdx.x;
    if (e < ETOT) coef[e] = dis[src[e]] * ew[e] * dis[dst[e]];
}

// agg[i,:] = dis[i]^2 * xw[i,:] + bias   (block per node, F threads)
__global__ void agg_init_kernel(const float* __restrict__ xw, const float* __restrict__ dis,
                                const float* __restrict__ bias, float* __restrict__ agg, int F) {
    int i = blockIdx.x, f = threadIdx.x;
    float dsq = dis[i] * dis[i];
    agg[(long)i * F + f] = dsq * xw[(long)i * F + f] + bias[f];
}

// agg[dst,:] += coef[e] * xw[src,:]   (block per edge, F threads, coalesced)
__global__ void edge_scatter_kernel(const int* __restrict__ src, const int* __restrict__ dst,
                                    const float* __restrict__ coef, const float* __restrict__ xw,
                                    float* __restrict__ agg, int F) {
    int e = blockIdx.x, f = threadIdx.x;
    int s = src[e], d = dst[e];
    float cf = coef[e];
    atomicAdd(&agg[(long)d * F + f], cf * xw[(long)s * F + f]);
}

__global__ void leaky_kernel(float* a, long n) {
    long i = (long)blockIdx.x * 256 + threadIdx.x;
    if (i < n) { float v = a[i]; a[i] = v >= 0.0f ? v : SLOPE * v; }
}

// ---------------- layer 3 (HID2 -> 1) ---------------------------------------
__global__ __launch_bounds__(256) void gemv_w3_kernel(const float* __restrict__ h2,
                                                      const float* __restrict__ W3,
                                                      float* __restrict__ xw3) {
    __shared__ float w[HID2];
    if (threadIdx.x < HID2) w[threadIdx.x] = W3[threadIdx.x];
    __syncthreads();
    int i = blockIdx.x * 256 + threadIdx.x;
    if (i >= NTOT) return;
    const float* h = h2 + (long)i * HID2;
    float acc = 0.0f;
#pragma unroll 8
    for (int k = 0; k < HID2; ++k) acc += h[k] * w[k];
    xw3[i] = acc;
}

__global__ void sc_init_kernel(const float* __restrict__ xw3, const float* __restrict__ dis,
                               const float* __restrict__ b3, float* __restrict__ sc) {
    int i = blockIdx.x * 256 + threadIdx.x;
    if (i < NTOT) sc[i] = dis[i] * dis[i] * xw3[i] + b3[0];
}

__global__ void scalar_scatter_kernel(const int* __restrict__ src, const int* __restrict__ dst,
                                      const float* __restrict__ coef,
                                      const float* __restrict__ xw3, float* __restrict__ sc) {
    int e = blockIdx.x * 256 + threadIdx.x;
    if (e < ETOT) atomicAdd(&sc[dst[e]], coef[e] * xw3[src[e]]);
}

__global__ void score_final_kernel(float* sc) {
    int i = blockIdx.x * 256 + threadIdx.x;
    if (i >= NTOT) return;
    float v = sc[i];
    v = v >= 0.0f ? v : SLOPE * v;
    if (isnan(v)) v = 0.0f;
    else if (isinf(v)) v = v > 0.0f ? 3.4028234663852886e38f : -3.4028234663852886e38f;
    sc[i] = v;
}

// ---------------- exact top-k via rank (reproduces lax.top_k ties) ----------
__global__ __launch_bounds__(512) void rank_sent_kernel(const float* __restrict__ score,
                                                        int* __restrict__ permI,
                                                        float* __restrict__ out_ss,
                                                        float* __restrict__ out_sb) {
    __shared__ float smem[BATCH * SENT];
    const int tid = threadIdx.x;           // 0..511
    const int b = tid >> 6, i = tid & 63;
    float si = score[(long)b * NNODE + i];
    smem[tid] = si;
    out_ss[tid] = si;                      // sentence_scores = s2[:, :S].reshape(-1)
    out_sb[tid] = (float)b;                // sentence_batch
    __syncthreads();
    int cnt = 0;
    for (int j = 0; j < SENT; ++j) {
        float sj = smem[b * SENT + j];
        cnt += (sj > si) || (sj == si && j < i);
    }
    if (cnt < KSENT) permI[b * KSENT + cnt] = b * NNODE + i;
}

__global__ __launch_bounds__(256) void rank_other_kernel(const float* __restrict__ score,
                                                         int* __restrict__ permI) {
    __shared__ float tile[256];
    const int b   = blockIdx.x >> 5;       // 32 blocks per batch
    const int blk = blockIdx.x & 31;
    const int i   = blk * 256 + threadIdx.x;
    const bool valid = (i < OTHERN);
    const float* s = score + (long)b * NNODE + SENT;
    float si = valid ? s[i] : 0.0f;
    int cnt = 0;
    for (int j0 = 0; j0 < OTHERN; j0 += 256) {
        int jl = j0 + threadIdx.x;
        __syncthreads();
        tile[threadIdx.x] = (jl < OTHERN) ? s[jl] : 0.0f;
        __syncthreads();
        int lim = min(256, OTHERN - j0);
        for (int jj = 0; jj < lim; ++jj) {
            float sj = tile[jj];
            int j = j0 + jj;
            cnt += (sj > si) || (sj == si && j < i);
        }
    }
    if (valid && cnt < KOTHER)
        permI[BATCH * KSENT + b * KOTHER + cnt] = b * NNODE + SENT + i;
}

// ---------------- nodemap + gathers -----------------------------------------
__global__ void nm_init_kernel(int* nm) {
    int i = blockIdx.x * 256 + threadIdx.x;
    if (i < NTOT) nm[i] = -1;
}

__global__ void nm_fill_kernel(const int* __restrict__ permI, int* __restrict__ nm) {
    int p = blockIdx.x * 256 + threadIdx.x;
    if (p < PTOT) nm[permI[p]] = p;
}

__global__ __launch_bounds__(256) void gather_nodes_kernel(const float* __restrict__ x,
    const float* __restrict__ score, const int* __restrict__ permI,
    float* __restrict__ xout, float* __restrict__ bout,
    float* __restrict__ pout, float* __restrict__ sout) {
    const int p = blockIdx.x;
    const int node = permI[p];
    const float s = score[node];
    const int f = threadIdx.x;
    xout[(long)p * CIN + f] = x[(long)node * CIN + f] * s;
    if (f == 0) {
        bout[p] = (float)(node / NNODE);
        pout[p] = (float)node;
        sout[p] = s;
    }
}

__global__ void edge_out_kernel(const int* __restrict__ src, const int* __restrict__ dst,
                                const float* __restrict__ ea, const int* __restrict__ nm,
                                float* __restrict__ eio, float* __restrict__ eao,
                                float* __restrict__ emo) {
    int e = blockIdx.x * 256 + threadIdx.x;
    if (e >= ETOT) return;
    int r = nm[src[e]], c = nm[dst[e]];
    bool m = (r >= 0) && (c >= 0);
    eio[e]        = m ? (float)r : -1.0f;
    eio[ETOT + e] = m ? (float)c : -1.0f;
    eao[e]        = m ? ea[e] : 0.0f;
    emo[e]        = m ? 1.0f : 0.0f;
}

// ---------------- driver -----------------------------------------------------
extern "C" void kernel_launch(void* const* d_in, const int* in_sizes, int n_in,
                              void* d_out, int out_size, void* d_ws, size_t ws_size,
                              hipStream_t stream) {
    const float* x  = (const float*)d_in[0];
    const int*   ei = (const int*)d_in[1];
    const float* ea = (const float*)d_in[2];
    const float* W1 = (const float*)d_in[6];
    const float* b1 = (const float*)d_in[7];
    const float* W2 = (const float*)d_in[8];
    const float* b2 = (const float*)d_in[9];
    const float* W3 = (const float*)d_in[10];
    const float* b3 = (const float*)d_in[11];
    const int* src = ei;
    const int* dst = ei + ETOT;

    char* ws = (char*)d_ws;
    float* dis   = (float*)ws; ws += (size_t)NTOT * 4;
    float* coef  = (float*)ws; ws += (size_t)ETOT * 4;
    float* bufA  = (float*)ws; ws += (size_t)NTOT * CIN * 4;   // xw1 / (xw2|agg2)
    float* bufB  = (float*)ws; ws += (size_t)NTOT * CIN * 4;   // agg1 -> h1
    float* xw3   = (float*)ws; ws += (size_t)NTOT * 4;
    float* sc    = (float*)ws; ws += (size_t)NTOT * 4;
    int*   permI = (int*)ws;   ws += (size_t)PTOT * 4;
    int*   nm    = (int*)ws;

    float* out = (float*)d_out;

    // degree normalization (shared across the 3 layers)
    zero_kernel<<<(NTOT + 255) / 256, 256, 0, stream>>>(dis, NTOT);
    deg_scatter_kernel<<<(ETOT + 255) / 256, 256, 0, stream>>>(dst, ea, dis);
    dis_kernel<<<(NTOT + 255) / 256, 256, 0, stream>>>(dis);
    coef_kernel<<<(ETOT + 255) / 256, 256, 0, stream>>>(src, dst, ea, dis, coef);

    // layer 1: xw1 = x @ W1 (WMMA bf16), agg + scatter + leaky
    gemm_bf16_wmma<<<dim3(NTOT / 128, HID / 64), 256, 0, stream>>>(x, W1, bufA, NTOT, HID, CIN);
    agg_init_kernel<<<NTOT, HID, 0, stream>>>(bufA, dis, b1, bufB, HID);
    edge_scatter_kernel<<<ETOT, HID, 0, stream>>>(src, dst, coef, bufA, bufB, HID);
    leaky_kernel<<<(int)(((long)NTOT * HID + 255) / 256), 256, 0, stream>>>(bufB, (long)NTOT * HID);

    // layer 2: xw2 = h1 @ W2 (WMMA bf16)
    gemm_bf16_wmma<<<dim3(NTOT / 128, HID2 / 64), 256, 0, stream>>>(bufB, W2, bufA, NTOT, HID2, HID);
    float* agg2 = bufA + (size_t)NTOT * HID2;
    agg_init_kernel<<<NTOT, HID2, 0, stream>>>(bufA, dis, b2, agg2, HID2);
    edge_scatter_kernel<<<ETOT, HID2, 0, stream>>>(src, dst, coef, bufA, agg2, HID2);
    leaky_kernel<<<(int)(((long)NTOT * HID2 + 255) / 256), 256, 0, stream>>>(agg2, (long)NTOT * HID2);

    // layer 3: scalar score
    gemv_w3_kernel<<<(NTOT + 255) / 256, 256, 0, stream>>>(agg2, W3, xw3);
    sc_init_kernel<<<(NTOT + 255) / 256, 256, 0, stream>>>(xw3, dis, b3, sc);
    scalar_scatter_kernel<<<(ETOT + 255) / 256, 256, 0, stream>>>(src, dst, coef, xw3, sc);
    score_final_kernel<<<(NTOT + 255) / 256, 256, 0, stream>>>(sc);

    // exact top-k (rank-based), perm, nodemap
    rank_sent_kernel<<<1, 512, 0, stream>>>(sc, permI, out + O_SS, out + O_SB);
    rank_other_kernel<<<BATCH * 32, 256, 0, stream>>>(sc, permI);
    nm_init_kernel<<<(NTOT + 255) / 256, 256, 0, stream>>>(nm);
    nm_fill_kernel<<<(PTOT + 255) / 256, 256, 0, stream>>>(permI, nm);

    // outputs
    gather_nodes_kernel<<<PTOT, 256, 0, stream>>>(x, sc, permI,
        out + O_X, out + O_BO, out + O_PM, out + O_SP);
    edge_out_kernel<<<(ETOT + 255) / 256, 256, 0, stream>>>(src, dst, ea, nm,
        out + O_EI, out + O_EA, out + O_EM);
}
